// MultiGaussKernel_15891378995447
// MI455X (gfx1250) — matrix operand order repaired and checked by reading, not compiled, not run on hardware
//
#include <hip/hip_runtime.h>
#include <hip/hip_bf16.h>

// ---------------------------------------------------------------------------
// MultiGauss attention for MI455X (gfx1250, wave32, WMMA).
//   B=2 H=8 L=M=2048 D=64.  out = 256MB fp32 -> HBM floor ~11us @ 23.3TB/s.
//   One bf16x3-split GEMM S=Q.K^T on v_wmma_f32_16x16x32_bf16, fused exp +
//   row-softmax-normalize so the 256MB output is written exactly once.
// ---------------------------------------------------------------------------

typedef __attribute__((ext_vector_type(16))) __bf16 v16bf;
typedef __attribute__((ext_vector_type(8)))  float  v8f;

#define NBH 16      // B*H
#define NL  2048    // L == M
#define ND  64      // head dim
#define A_SC (-0.0625f)   // -0.5 / sqrt(64)

struct __align__(16) U4 { unsigned int x, y, z, w; };
union FragU { U4 u[2]; v16bf v; };   // 32B: one 16x32-bf16 WMMA operand per lane

// ---------------------------------------------------------------------------
// Prologue: per (bh,i) compute rowP/colP terms and bf16 hi/lo splits of q,k.
//   colP_c[m] = a*|k_m - mu_c|^2 - 1e6*(1-mask[b,m])
//   rowP_c[l] = a*(|q_l|^2 + 2*q_l.mu_c)
//   q = qh + ql (bf16 truncation split), same for k  -> bf16x3 GEMM later.
// ---------------------------------------------------------------------------
__global__ __launch_bounds__(256) void mg_pre(
    const float* __restrict__ q, const float* __restrict__ k,
    const float* __restrict__ mask, const float* __restrict__ mu,
    float* __restrict__ colP0, float* __restrict__ colP1,
    float* __restrict__ rowP0, float* __restrict__ rowP1,
    unsigned short* __restrict__ qhi, unsigned short* __restrict__ qlo,
    unsigned short* __restrict__ khi, unsigned short* __restrict__ klo)
{
  int idx = blockIdx.x * blockDim.x + threadIdx.x;   // 0 .. NBH*NL-1
  int bh = idx >> 11;
  int i  = idx & (NL - 1);
  int b  = bh >> 3;
  int h  = bh & 7;

  const float* qp  = q  + (size_t)idx * ND;
  const float* kp  = k  + (size_t)idx * ND;
  const float* u0p = mu + h * ND;             // mu[0,h,:]
  const float* u1p = mu + 8 * ND + h * ND;    // mu[1,h,:]

  unsigned int* qhp = (unsigned int*)(qhi + (size_t)idx * ND);
  unsigned int* qlp = (unsigned int*)(qlo + (size_t)idx * ND);
  unsigned int* khp = (unsigned int*)(khi + (size_t)idx * ND);
  unsigned int* klp = (unsigned int*)(klo + (size_t)idx * ND);

  float ak0 = 0.f, ak1 = 0.f, aq = 0.f, qm0 = 0.f, qm1 = 0.f;

  #pragma unroll 4
  for (int d = 0; d < ND; d += 2) {
    float qv0 = qp[d],   qv1 = qp[d + 1];
    float kv0 = kp[d],   kv1 = kp[d + 1];
    float u00 = u0p[d],  u01 = u0p[d + 1];
    float u10 = u1p[d],  u11 = u1p[d + 1];

    float t;
    t = kv0 - u00; ak0 = fmaf(t, t, ak0);
    t = kv1 - u01; ak0 = fmaf(t, t, ak0);
    t = kv0 - u10; ak1 = fmaf(t, t, ak1);
    t = kv1 - u11; ak1 = fmaf(t, t, ak1);
    aq  = fmaf(qv0, qv0, fmaf(qv1, qv1, aq));
    qm0 = fmaf(qv0, u00, fmaf(qv1, u01, qm0));
    qm1 = fmaf(qv0, u10, fmaf(qv1, u11, qm1));

    // truncation split: f = hi(bf16) + lo, lo stored as truncated bf16 too
    unsigned int qb0 = __float_as_uint(qv0), qb1 = __float_as_uint(qv1);
    unsigned int kb0 = __float_as_uint(kv0), kb1 = __float_as_uint(kv1);
    float qr0 = qv0 - __uint_as_float(qb0 & 0xFFFF0000u);
    float qr1 = qv1 - __uint_as_float(qb1 & 0xFFFF0000u);
    float kr0 = kv0 - __uint_as_float(kb0 & 0xFFFF0000u);
    float kr1 = kv1 - __uint_as_float(kb1 & 0xFFFF0000u);
    qhp[d >> 1] = (qb0 >> 16) | (qb1 & 0xFFFF0000u);
    khp[d >> 1] = (kb0 >> 16) | (kb1 & 0xFFFF0000u);
    qlp[d >> 1] = (__float_as_uint(qr0) >> 16) | (__float_as_uint(qr1) & 0xFFFF0000u);
    klp[d >> 1] = (__float_as_uint(kr0) >> 16) | (__float_as_uint(kr1) & 0xFFFF0000u);
  }

  float negv = 1.0e6f * (1.0f - mask[b * NL + i]);
  colP0[idx] = A_SC * ak0 - negv;
  colP1[idx] = A_SC * ak1 - negv;
  rowP0[idx] = A_SC * (aq + 2.0f * qm0);
  rowP1[idx] = A_SC * (aq + 2.0f * qm1);
}

// ---------------------------------------------------------------------------
// Main: one WG (16 waves) = 16-row strip x all 2048 cols of one (b,h).
// Each wave owns 8 column tiles of 16 (128 cols) kept in v8f accumulators.
// bf16x3 GEMM -> fused exp -> shfl+LDS rowsum -> normalize -> single write.
// ---------------------------------------------------------------------------
__global__ __launch_bounds__(512, 1) void mg_main(
    const float* __restrict__ pi,
    const float* __restrict__ colP0, const float* __restrict__ colP1,
    const float* __restrict__ rowP0, const float* __restrict__ rowP1,
    const unsigned short* __restrict__ qhi, const unsigned short* __restrict__ qlo,
    const unsigned short* __restrict__ khi, const unsigned short* __restrict__ klo,
    float* __restrict__ out)
{
  __shared__ float lds_rsum[16];

  const int bh  = blockIdx.x >> 7;
  const int l0  = (blockIdx.x & 127) << 4;
  const int tid = threadIdx.x;
  const int wave = tid >> 5;
  const int lane = tid & 31;
  const int n  = lane & 15;   // N / row index inside 16-lane half
  const int hh = lane >> 4;   // lane half (selects K-range / M-offset)

  if (tid < 16) lds_rsum[tid] = 0.0f;
  __syncthreads();

  const float pic0 = fminf(fmaxf(pi[0], 0.f), 1.f);
  const float pic1 = fminf(fmaxf(pi[1], 0.f), 1.f);

  // --- A fragments (Q strip rows l0..l0+15), resident for whole kernel ---
  // A 16x32 bf16 layout: lane(row=n, half=hh) holds K = {8hh..8hh+7} U {16+8hh..+7}
  const size_t qrow = ((size_t)bh * NL + l0 + n) * ND;
  FragU qh[2], ql[2];
  #pragma unroll
  for (int c = 0; c < 2; ++c) {
    const int d0 = 32 * c + 8 * hh;
    qh[c].u[0] = *(const U4*)(qhi + qrow + d0);
    qh[c].u[1] = *(const U4*)(qhi + qrow + d0 + 16);
    ql[c].u[0] = *(const U4*)(qlo + qrow + d0);
    ql[c].u[1] = *(const U4*)(qlo + qrow + d0 + 16);
  }

  v8f acc[8];
  #pragma unroll
  for (int t = 0; t < 8; ++t) acc[t] = (v8f){0.f,0.f,0.f,0.f,0.f,0.f,0.f,0.f};

  const int mbase = wave << 7;       // this wave's first column (wave*128)

  #pragma unroll
  for (int t = 0; t < 8; ++t) {
    const int m0 = mbase + (t << 4);
    // B 32x16 bf16 layout: lane(col=n, half=hh) holds K = {8hh..8hh+7} U {16+8hh..+7}
    const size_t krow = ((size_t)bh * NL + m0 + n) * ND;
    #pragma unroll
    for (int c = 0; c < 2; ++c) {
      const int d0 = 32 * c + 8 * hh;
      FragU kh, kl;
      kh.u[0] = *(const U4*)(khi + krow + d0);
      kh.u[1] = *(const U4*)(khi + krow + d0 + 16);
      kl.u[0] = *(const U4*)(klo + krow + d0);
      kl.u[1] = *(const U4*)(klo + krow + d0 + 16);
      // bf16x3: qh*kh + ql*kh + qh*kl  (drop ql*kl, ~2^-16 rel error)
      acc[t] = __builtin_amdgcn_wmma_f32_16x16x32_bf16(
                 false, qh[c].v, false, kh.v, (short)0, acc[t], false, false);
      acc[t] = __builtin_amdgcn_wmma_f32_16x16x32_bf16(
                 false, ql[c].v, false, kh.v, (short)0, acc[t], false, false);
      acc[t] = __builtin_amdgcn_wmma_f32_16x16x32_bf16(
                 false, qh[c].v, false, kl.v, (short)0, acc[t], false, false);
    }
  }

  // --- epilogue: S -> unnormalized attn (in place), accumulate row sums ---
  // C/D layout: lane(n,hh), vgpr v -> row (v + 8hh), col n of the tile.
  float rp0[8], rp1[8];
  #pragma unroll
  for (int v = 0; v < 8; ++v) {
    rp0[v] = rowP0[bh * NL + l0 + v + 8 * hh];
    rp1[v] = rowP1[bh * NL + l0 + v + 8 * hh];
  }

  float rsum[8];
  #pragma unroll
  for (int v = 0; v < 8; ++v) rsum[v] = 0.f;

  #pragma unroll
  for (int t = 0; t < 8; ++t) {
    const int m = mbase + (t << 4) + n;
    const float c0 = colP0[bh * NL + m];
    const float c1 = colP1[bh * NL + m];
    #pragma unroll
    for (int v = 0; v < 8; ++v) {
      const float s = 0.125f * acc[t][v];          // -2a = +0.125
      const float e = pic0 * __expf(rp0[v] + c0 + s)
                    + pic1 * __expf(rp1[v] + c1 + s);
      acc[t][v] = e;
      rsum[v] += e;
    }
  }

  // reduce over the 16 lanes of each half (rows are disjoint across halves)
  #pragma unroll
  for (int v = 0; v < 8; ++v) {
    float x = rsum[v];
    x += __shfl_xor(x, 1, 32);
    x += __shfl_xor(x, 2, 32);
    x += __shfl_xor(x, 4, 32);
    x += __shfl_xor(x, 8, 32);
    rsum[v] = x;
  }
  if (n == 0) {
    #pragma unroll
    for (int v = 0; v < 8; ++v) atomicAdd(&lds_rsum[v + 8 * hh], rsum[v]);
  }
  __syncthreads();

  float inv[8];
  #pragma unroll
  for (int v = 0; v < 8; ++v) inv[v] = __builtin_amdgcn_rcpf(lds_rsum[v + 8 * hh]);

  float* ob = out + (size_t)bh * NL * NL;
  #pragma unroll
  for (int t = 0; t < 8; ++t) {
    const int m = mbase + (t << 4) + n;
    #pragma unroll
    for (int v = 0; v < 8; ++v) {
      ob[(size_t)(l0 + v + 8 * hh) * NL + m] = acc[t][v] * inv[v];
    }
  }
}

// ---------------------------------------------------------------------------
extern "C" void kernel_launch(void* const* d_in, const int* in_sizes, int n_in,
                              void* d_out, int out_size, void* d_ws, size_t ws_size,
                              hipStream_t stream) {
  const float* q    = (const float*)d_in[0];
  const float* key  = (const float*)d_in[1];
  const float* mask = (const float*)d_in[2];
  const float* mu   = (const float*)d_in[3];
  const float* pi   = (const float*)d_in[4];
  float* out = (float*)d_out;

  // workspace layout (needs 512KB terms + 16MB bf16 splits = 16.5MB)
  char* ws = (char*)d_ws;
  float* colP0 = (float*)ws;
  float* colP1 = colP0 + (size_t)NBH * NL;
  float* rowP0 = colP1 + (size_t)NBH * NL;
  float* rowP1 = rowP0 + (size_t)NBH * NL;
  unsigned short* qhi = (unsigned short*)(rowP1 + (size_t)NBH * NL);
  unsigned short* qlo = qhi + (size_t)NBH * NL * ND;
  unsigned short* khi = qlo + (size_t)NBH * NL * ND;
  unsigned short* klo = khi + (size_t)NBH * NL * ND;

  mg_pre<<<(NBH * NL) / 256, 256, 0, stream>>>(
      q, key, mask, mu, colP0, colP1, rowP0, rowP1, qhi, qlo, khi, klo);

  mg_main<<<NBH * (NL / 16), 512, 0, stream>>>(
      pi, colP0, colP1, rowP0, rowP1, qhi, qlo, khi, klo, out);
}
